// ConvMultiheadAttention_32796370272272
// MI455X (gfx1250) — matrix operand order repaired and checked by reading, not compiled, not run on hardware
//
#include <hip/hip_runtime.h>
#include <hip/hip_bf16.h>

typedef __attribute__((ext_vector_type(16))) __bf16 v16bf;
typedef __attribute__((ext_vector_type(8)))  __bf16 v8bf;
typedef __attribute__((ext_vector_type(8)))  float  v8f;
typedef __attribute__((ext_vector_type(4)))  int    v4i;

union V16 { v16bf v; v8bf h[2]; };

#define B_  8
#define C_  512
#define L_  2048
#define LP_ (L_ + 2)   // padded rows: [0]=zeros, [1..L]=data, [L+1]=zeros
#define H_  8
#define DH_ 64
static constexpr float SCALE = 0.04419417382415922f; // 1/sqrt(512)

#if defined(__HIP_DEVICE_COMPILE__) && __has_builtin(__builtin_amdgcn_global_load_async_to_lds_b128)
#define HAS_ASYNC_LDS 1
#else
#define HAS_ASYNC_LDS 0
#endif

#if defined(__HIP_DEVICE_COMPILE__) && __has_builtin(__builtin_amdgcn_s_wait_asynccnt)
#define WAIT_ASYNC(N) __builtin_amdgcn_s_wait_asynccnt(N)
#else
#define WAIT_ASYNC(N) asm volatile("s_wait_asynccnt %0" :: "i"(N) : "memory")
#endif

__device__ __forceinline__ v8bf cvt8(v8f a) {
  v8bf r;
  #pragma unroll
  for (int i = 0; i < 8; ++i) r[i] = (__bf16)a[i];
  return r;
}

// ---------------------------------------------------------------------------
// Kernel 1: transpose+convert x[B,C,L] f32 -> xbTp[B, 1+L, C] bf16 (LDS tile)
// ---------------------------------------------------------------------------
__global__ __launch_bounds__(256) void k_xpose(const float* __restrict__ x,
                                               __bf16* __restrict__ xbTp) {
  __shared__ float tile[32][33];
  const int b  = blockIdx.z;
  const int l0 = blockIdx.x * 32, c0 = blockIdx.y * 32;
  const int tx = threadIdx.x, ty = threadIdx.y; // (32,8)
  const float* xb = x + (size_t)b * C_ * L_;
  #pragma unroll
  for (int i = 0; i < 4; ++i)
    tile[ty + 8 * i][tx] = xb[(size_t)(c0 + ty + 8 * i) * L_ + l0 + tx];
  __syncthreads();
  __bf16* o = xbTp + (size_t)b * LP_ * C_ + C_;   // +C_ : skip zero pad row
  #pragma unroll
  for (int i = 0; i < 4; ++i)
    o[(size_t)(l0 + ty + 8 * i) * C_ + c0 + tx] = (__bf16)tile[tx][ty + 8 * i];
}

// Zero the two pad rows per batch (workspace is poisoned by the harness).
__global__ __launch_bounds__(256) void k_zpad(__bf16* __restrict__ xbTp) {
  const int idx = blockIdx.x * 256 + threadIdx.x;   // 8 * 2 * 512 = 8192 exact
  const int b = idx >> 10;
  const int r = (idx >> 9) & 1;                     // 0 -> row 0, 1 -> row L+1
  const int c = idx & (C_ - 1);
  xbTp[((size_t)b * LP_ + (r ? (L_ + 1) : 0)) * C_ + c] = (__bf16)0.0f;
}

// ---------------------------------------------------------------------------
// Kernel 2: repack conv weights -> wb[p][tap][cout][cin] bf16
// ---------------------------------------------------------------------------
__global__ __launch_bounds__(256) void k_wcvt(const float* __restrict__ w0,
                                              const float* __restrict__ w1,
                                              const float* __restrict__ w2,
                                              __bf16* __restrict__ wb) {
  const int idx = blockIdx.x * 256 + threadIdx.x;   // exactly 3*512*512*3
  const int p   = idx / (C_ * C_ * 3);
  const int r0  = idx % (C_ * C_ * 3);
  const int co  = r0 / (C_ * 3);
  const int r1  = r0 % (C_ * 3);
  const int ci  = r1 / 3;
  const int tap = r1 % 3;
  const float* w = (p == 0) ? w0 : (p == 1) ? w1 : w2;
  wb[(((size_t)p * 3 + tap) * C_ + co) * C_ + ci] =
      (__bf16)w[((size_t)co * C_ + ci) * 3 + tap];
}

// ---------------------------------------------------------------------------
// Kernel 3: q/k/v conv projections as implicit GEMM with WMMA bf16.
// Fully-unrolled async double-buffered LDS pipeline: 24 stages of K=64
// (3 taps x 8 cin-blocks); 2 async b128/thread/stage; s_wait_asynccnt<=2
// keeps one stage in flight; 8 wmma + next-stage A-prefetch per barrier pair.
//   q -> qT[b,h,l,d]*SCALE   k -> kT[b,h,l,d]   v -> vb[b,h,d,l]
// ---------------------------------------------------------------------------
__global__ __launch_bounds__(256) void k_conv(const __bf16* __restrict__ xbTp,
                                              const __bf16* __restrict__ wb,
                                              const float* __restrict__ bias0,
                                              const float* __restrict__ bias1,
                                              const float* __restrict__ bias2,
                                              __bf16* __restrict__ qT,
                                              __bf16* __restrict__ kT,
                                              __bf16* __restrict__ vb) {
  const int lane = threadIdx.x & 31, wave = threadIdx.x >> 5;
  const int ln = lane & 15, hi = lane >> 4;
  const int b = blockIdx.z / 3, p = blockIdx.z % 3;
  const int c0 = blockIdx.y * 128 + wave * 16;   // cout tile (8 M-waves)
  const int l0 = blockIdx.x * 64;                // shared l-range, 4 n-tiles

  v8f acc[4] = {};
  const __bf16* xbase = xbTp + (size_t)b * LP_ * C_;

#if HAS_ASYNC_LDS
  // ---- async global->LDS pipeline -----------------------------------------
  __shared__ __align__(16) __bf16 Xs[2][64][72];   // 144B row stride, 18 KB
  const int srow = threadIdx.x >> 2;               // 0..63  (l-row in tile)
  const int scol = (threadIdx.x & 3) * 8;          // 16B chunk within 64 cin

  // stage s global addr = gstream + s*64 elements (tap*C_ + cb64 == s*64)
  const __bf16* gstream = xbase + (size_t)(l0 + srow) * C_ + scol;

  auto stage = [&](int s) {
    const __bf16* gp = gstream + (size_t)s * 64;
    __bf16* lp = &Xs[s & 1][srow][scol];
    __builtin_amdgcn_global_load_async_to_lds_b128(
        (__attribute__((address_space(1))) v4i*)gp,
        (__attribute__((address_space(3))) v4i*)lp, 0, 0);
    __builtin_amdgcn_global_load_async_to_lds_b128(
        (__attribute__((address_space(1))) v4i*)(gp + 32),
        (__attribute__((address_space(3))) v4i*)(lp + 32), 0, 0);
  };

  const __bf16* wlane = wb + ((size_t)(p * 3) * C_ + (c0 + ln)) * C_ + hi * 16;
  auto loadA = [&](V16& a, int s, int half) {
    const __bf16* wrow =
        wlane + (size_t)(s >> 3) * (C_ * C_) + ((s & 7) << 6) + half * 32;
    a.h[0] = *(const v8bf*)(wrow);
    a.h[1] = *(const v8bf*)(wrow + 8);
  };

  stage(0);
  V16 a0, a1;
  loadA(a0, 0, 0);
  loadA(a1, 0, 1);
  #pragma unroll
  for (int s = 0; s < 24; ++s) {
    __syncthreads();                   // all waves done reading buf[(s+1)&1]
    if (s + 1 < 24) {
      stage(s + 1);
      WAIT_ASYNC(2);                   // stage s landed (in-order)
    } else {
      WAIT_ASYNC(0);
    }
    __syncthreads();                   // all waves' chunks of stage s visible

    // prefetch next stage's A fragments; overlaps this stage's 8 wmmas
    V16 na0, na1;
    if (s + 1 < 24) { loadA(na0, s + 1, 0); loadA(na1, s + 1, 1); }

    const __bf16* xr = &Xs[s & 1][ln][hi * 16];
    #pragma unroll
    for (int nt = 0; nt < 4; ++nt) {
      V16 bfr;
      bfr.h[0] = *(const v8bf*)(xr + nt * 16 * 72);
      bfr.h[1] = *(const v8bf*)(xr + nt * 16 * 72 + 8);
      acc[nt] = __builtin_amdgcn_wmma_f32_16x16x32_bf16(
          false, a0.v, false, bfr.v, (short)0, acc[nt], false, false);
    }
    #pragma unroll
    for (int nt = 0; nt < 4; ++nt) {
      V16 bfr;
      bfr.h[0] = *(const v8bf*)(xr + nt * 16 * 72 + 32);
      bfr.h[1] = *(const v8bf*)(xr + nt * 16 * 72 + 40);
      acc[nt] = __builtin_amdgcn_wmma_f32_16x16x32_bf16(
          false, a1.v, false, bfr.v, (short)0, acc[nt], false, false);
    }
    a0 = na0;                          // free renaming (fully unrolled)
    a1 = na1;
  }
#else
  // ---- fallback: direct global loads, ping-pong fragments -----------------
#define LOADF(A, Bv, CB)                                                     \
  do {                                                                       \
    (A).h[0] = *(const v8bf*)(wrow + (CB));                                  \
    (A).h[1] = *(const v8bf*)(wrow + (CB) + 8);                              \
    _Pragma("unroll")                                                        \
    for (int nt = 0; nt < 4; ++nt)                                           \
      (Bv)[nt] = *(const v16bf*)(xrow + (CB) + nt * 16 * C_);                \
  } while (0)
#define WMMA4(A, Bv)                                                         \
  do {                                                                       \
    _Pragma("unroll")                                                        \
    for (int nt = 0; nt < 4; ++nt)                                           \
      acc[nt] = __builtin_amdgcn_wmma_f32_16x16x32_bf16(                     \
          false, (A).v, false, (Bv)[nt], (short)0, acc[nt], false, false);   \
  } while (0)
  #pragma unroll
  for (int tap = 0; tap < 3; ++tap) {
    const __bf16* wrow = wb + (((size_t)(p * 3 + tap)) * C_ + (c0 + ln)) * C_ + hi * 16;
    const __bf16* xrow = xbase + (size_t)(l0 + tap + ln) * C_ + hi * 16;
    V16 a0, a1;
    v16bf b0f[4], b1f[4];
    LOADF(a0, b0f, 0);
    #pragma unroll
    for (int cb = 0; cb < C_; cb += 64) {
      LOADF(a1, b1f, cb + 32);
      WMMA4(a0, b0f);
      if (cb + 64 < C_) LOADF(a0, b0f, cb + 64);
      WMMA4(a1, b1f);
    }
  }
#undef LOADF
#undef WMMA4
#endif

  // bias (per cout row: m = r + 8*hi)
  const float* bias = (p == 0) ? bias0 : (p == 1) ? bias1 : bias2;
  v8f bv;
  {
    const float4 f0 = *(const float4*)(bias + c0 + hi * 8);
    const float4 f1 = *(const float4*)(bias + c0 + hi * 8 + 4);
    bv[0] = f0.x; bv[1] = f0.y; bv[2] = f0.z; bv[3] = f0.w;
    bv[4] = f1.x; bv[5] = f1.y; bv[6] = f1.z; bv[7] = f1.w;
  }
  #pragma unroll
  for (int nt = 0; nt < 4; ++nt) acc[nt] = acc[nt] + bv;

  const int h  = c0 >> 6;          // head (uniform per wave)
  const int d0 = c0 & 63;          // d base within head
  const size_t bh = (size_t)b * H_ + h;
  if (p == 2) {
    // v -> [b,h,d,l] (B-operand layout for P @ V^T)
    #pragma unroll
    for (int nt = 0; nt < 4; ++nt) {
      const int l = l0 + nt * 16 + ln;
      #pragma unroll
      for (int r = 0; r < 8; ++r) {
        const int d = d0 + hi * 8 + r;
        vb[(bh * DH_ + d) * L_ + l] = (__bf16)acc[nt][r];
      }
    }
  } else {
    __bf16* dst = (p == 0) ? qT : kT;
    const float s = (p == 0) ? SCALE : 1.0f;
    #pragma unroll
    for (int nt = 0; nt < 4; ++nt) {
      const int l = l0 + nt * 16 + ln;
      v8f t = acc[nt] * s;
      *(v8bf*)(dst + (bh * L_ + l) * DH_ + d0 + hi * 8) = cvt8(t);
    }
  }
}

// ---------------------------------------------------------------------------
// Kernel 4: flash attention. 8 waves/WG, 16 queries per wave, key blocks of 32.
// V fragments load before the softmax so they are in flight during the whole
// VALU chain + LDS round-trip.
// ---------------------------------------------------------------------------
__global__ __launch_bounds__(256) void k_attn(const __bf16* __restrict__ qT,
                                              const __bf16* __restrict__ kT,
                                              const __bf16* __restrict__ vb,
                                              float* __restrict__ out) {
  __shared__ __align__(16) __bf16 Plds[8][16][32]; // per-wave P staging (1 KB)
  const int lane = threadIdx.x & 31, wave = threadIdx.x >> 5;
  const int ln = lane & 15, hi = lane >> 4;
  const int b = blockIdx.z, h = blockIdx.y;
  const int q0 = blockIdx.x * 128 + wave * 16;
  const size_t head = ((size_t)b * H_ + h) * (size_t)L_ * DH_;

  V16 qa[2];
  #pragma unroll
  for (int c = 0; c < 2; ++c) {
    const __bf16* qrow = qT + head + (size_t)(q0 + ln) * DH_ + c * 32 + hi * 16;
    qa[c].h[0] = *(const v8bf*)(qrow);
    qa[c].h[1] = *(const v8bf*)(qrow + 8);
  }

  v8f oacc[4] = {};
  v8f mrow, lrow;
  #pragma unroll
  for (int r = 0; r < 8; ++r) { mrow[r] = -1e30f; lrow[r] = 0.0f; }

  const __bf16* kbase = kT + head;
  const __bf16* vbase = vb + head;

  for (int kb = 0; kb < L_; kb += 32) {
    const int kbn = (kb + 32) & (L_ - 1);
    __builtin_prefetch(kbase + (size_t)(kbn + lane) * DH_, 0, 1);
    __builtin_prefetch(vbase + (size_t)(lane << 1) * L_ + kbn, 0, 1);

    v16bf kf[4];
    #pragma unroll
    for (int c = 0; c < 2; ++c) {
      kf[c * 2 + 0] = *(const v16bf*)(kbase + (size_t)(kb + ln) * DH_ + c * 32 + hi * 16);
      kf[c * 2 + 1] = *(const v16bf*)(kbase + (size_t)(kb + 16 + ln) * DH_ + c * 32 + hi * 16);
    }
    v16bf vf[4];
    #pragma unroll
    for (int dt = 0; dt < 4; ++dt)
      vf[dt] = *(const v16bf*)(vbase + (size_t)(dt * 16 + ln) * L_ + kb + hi * 16);

    v8f s0 = {}, s1 = {};
    #pragma unroll
    for (int c = 0; c < 2; ++c) {
      s0 = __builtin_amdgcn_wmma_f32_16x16x32_bf16(false, qa[c].v, false, kf[c * 2 + 0], (short)0, s0, false, false);
      s1 = __builtin_amdgcn_wmma_f32_16x16x32_bf16(false, qa[c].v, false, kf[c * 2 + 1], (short)0, s1, false, false);
    }

    v8f alpha;
    #pragma unroll
    for (int r = 0; r < 8; ++r) {
      float t = fmaxf(s0[r], s1[r]);
      t = fmaxf(t, __shfl_xor(t, 1, 16));
      t = fmaxf(t, __shfl_xor(t, 2, 16));
      t = fmaxf(t, __shfl_xor(t, 4, 16));
      t = fmaxf(t, __shfl_xor(t, 8, 16));
      const float mn = fmaxf(mrow[r], t);
      alpha[r] = __expf(mrow[r] - mn);
      const float p0 = __expf(s0[r] - mn);
      const float p1 = __expf(s1[r] - mn);
      s0[r] = p0; s1[r] = p1;
      float rs = p0 + p1;
      rs += __shfl_xor(rs, 1, 16);
      rs += __shfl_xor(rs, 2, 16);
      rs += __shfl_xor(rs, 4, 16);
      rs += __shfl_xor(rs, 8, 16);
      lrow[r] = alpha[r] * lrow[r] + rs;
      mrow[r] = mn;
    }
    #pragma unroll
    for (int dt = 0; dt < 4; ++dt) oacc[dt] = oacc[dt] * alpha;

    #pragma unroll
    for (int r = 0; r < 8; ++r) {
      Plds[wave][r + 8 * hi][ln]      = (__bf16)s0[r];
      Plds[wave][r + 8 * hi][16 + ln] = (__bf16)s1[r];
    }
    V16 pa;
    pa.h[0] = *(const v8bf*)(&Plds[wave][ln][hi * 16]);
    pa.h[1] = *(const v8bf*)(&Plds[wave][ln][hi * 16 + 8]);

    #pragma unroll
    for (int dt = 0; dt < 4; ++dt)
      oacc[dt] = __builtin_amdgcn_wmma_f32_16x16x32_bf16(false, pa.v, false, vf[dt], (short)0, oacc[dt], false, false);
  }

  #pragma unroll
  for (int dt = 0; dt < 4; ++dt) {
    float vals[8];
    #pragma unroll
    for (int r = 0; r < 8; ++r) vals[r] = oacc[dt][r] / lrow[r];
    float* op = out + ((size_t)b * C_ + h * 64 + dt * 16 + ln) * L_ + q0 + hi * 8;
    *(float4*)(op)     = make_float4(vals[0], vals[1], vals[2], vals[3]);
    *(float4*)(op + 4) = make_float4(vals[4], vals[5], vals[6], vals[7]);
  }
}

// ---------------------------------------------------------------------------
extern "C" void kernel_launch(void* const* d_in, const int* in_sizes, int n_in,
                              void* d_out, int out_size, void* d_ws, size_t ws_size,
                              hipStream_t stream) {
  (void)in_sizes; (void)n_in; (void)out_size; (void)ws_size;
  const float* x  = (const float*)d_in[0];
  const float* w0 = (const float*)d_in[1];
  const float* b0 = (const float*)d_in[2];
  const float* w1 = (const float*)d_in[3];
  const float* b1 = (const float*)d_in[4];
  const float* w2 = (const float*)d_in[5];
  const float* b2 = (const float*)d_in[6];
  float* out = (float*)d_out;

  char* ws = (char*)d_ws;
  __bf16* xbTp = (__bf16*)(ws);              // 16.02 MB : padded x^T bf16 [B,L+2,C]
  __bf16* wb   = (__bf16*)(ws + 16793600);   // 4.5 MB   : weights bf16 [p,tap,co,ci]
  __bf16* qT   = (__bf16*)(ws + 21512192);   // 16 MB    : q bf16 [b,h,l,d] (pre-scaled)
  __bf16* kT   = (__bf16*)(ws + 38289408);   // 16 MB    : k bf16 [b,h,l,d]
  __bf16* vb   = (__bf16*)(ws + 55066624);   // 16 MB    : v bf16 [b,h,d,l]

  k_zpad <<<dim3(32), dim3(256), 0, stream>>>(xbTp);
  k_xpose<<<dim3(L_ / 32, C_ / 32, B_), dim3(32, 8), 0, stream>>>(x, xbTp);
  k_wcvt <<<dim3((3 * C_ * C_ * 3) / 256), dim3(256), 0, stream>>>(w0, w1, w2, wb);
  k_conv <<<dim3(L_ / 64, C_ / 128, B_ * 3), dim3(256), 0, stream>>>(xbTp, wb, b0, b1, b2, qT, kT, vb);
  k_attn <<<dim3(L_ / 128, H_, B_), dim3(256), 0, stream>>>(qT, kT, vb, out);
}
